// TitanLongTermMemory_39986145526017
// MI455X (gfx1250) — matrix operand order repaired and checked by reading, not compiled, not run on hardware
//
#include <hip/hip_runtime.h>

// ---------------------------------------------------------------------------
// Types for CDNA5 WMMA (gfx1250, wave32)
// ---------------------------------------------------------------------------
typedef __attribute__((ext_vector_type(16))) __bf16 bf16x16;
typedef __attribute__((ext_vector_type(8)))  float  f32x8;

union FragBF16 { uint4 q[2]; bf16x16 v; };

__device__ __forceinline__ __bf16 to_bf16(float f) {
  unsigned u = __float_as_uint(f);
  u = (u + 0x7FFFu + ((u >> 16) & 1u)) >> 16;   // round-to-nearest-even
  unsigned short h = (unsigned short)u;
  return __builtin_bit_cast(__bf16, h);
}

// CDNA5 async DMA: global -> LDS, 16B per lane, tracked by ASYNCcnt.
// lds_off = LDS byte address (low 32 bits of the flat shared-aperture address).
__device__ __forceinline__ void async_load_b128(unsigned lds_off, const void* gaddr) {
  asm volatile("global_load_async_to_lds_b128 %0, %1, off"
               :: "v"(lds_off), "v"((unsigned long long)(size_t)gaddr)
               : "memory");
}
#define WAIT_ASYNC(n) asm volatile("s_wait_asynccnt " #n ::: "memory")

// ---------------------------------------------------------------------------
// f32 -> bf16 conversion (grid-stride)
// ---------------------------------------------------------------------------
__global__ void cvt_f32_bf16_k(const float* __restrict__ in, __bf16* __restrict__ out, long n) {
  for (long i = (long)blockIdx.x * blockDim.x + threadIdx.x; i < n;
       i += (long)gridDim.x * blockDim.x)
    out[i] = to_bf16(in[i]);
}

// ---------------------------------------------------------------------------
// GEMM: C[M,N] = A[M,K](bf16) @ W[N,K]^T(bf16) + bias, act: 0=none 1=silu 2=sigmoid
// Block: 256 threads = 8 waves; tile 128x128, BK=32, double-buffered LDS fed by
// GLOBAL_LOAD_ASYNC_TO_LDS_B128 so next-tile DMA overlaps WMMA on current tile.
// Wave (wm in 0..3, wn in 0..1) computes 32x64 via 2x4 WMMA 16x16 tiles.
// ---------------------------------------------------------------------------
__global__ __launch_bounds__(256) void gemm_bf16_wmma_k(
    const __bf16* __restrict__ A, const __bf16* __restrict__ W,
    const float* __restrict__ bias, float* __restrict__ Cf, __bf16* __restrict__ Cb,
    int M, int N, int K, int act)
{
  __shared__ __attribute__((aligned(16))) __bf16 As[2][128 * 32];
  __shared__ __attribute__((aligned(16))) __bf16 Bs[2][128 * 32];

  const int tid  = threadIdx.x;
  const int m0   = blockIdx.x * 128;
  const int n0   = blockIdx.y * 128;
  const int wave = tid >> 5, lane = tid & 31;
  const int wm   = wave >> 1, wn = wave & 1;
  const int lr   = lane & 15, kh = lane >> 4;

  // per-thread DMA assignment: 2 chunks (16B) each for A and W per tile
  const int c0 = tid, c1 = tid + 256;
  const int r0 = c0 >> 2, kc0 = (c0 & 3) << 3;
  const int r1 = c1 >> 2, kc1 = (c1 & 3) << 3;
  const __bf16* gA0 = A + (size_t)(m0 + r0) * K + kc0;
  const __bf16* gA1 = A + (size_t)(m0 + r1) * K + kc1;
  const __bf16* gW0 = W + (size_t)(n0 + r0) * K + kc0;
  const __bf16* gW1 = W + (size_t)(n0 + r1) * K + kc1;
  unsigned oA0[2], oA1[2], oW0[2], oW1[2];
#pragma unroll
  for (int b = 0; b < 2; ++b) {
    oA0[b] = (unsigned)(size_t)&As[b][r0 * 32 + kc0];
    oA1[b] = (unsigned)(size_t)&As[b][r1 * 32 + kc1];
    oW0[b] = (unsigned)(size_t)&Bs[b][r0 * 32 + kc0];
    oW1[b] = (unsigned)(size_t)&Bs[b][r1 * 32 + kc1];
  }
  auto issue = [&](int b, int kk) {
    async_load_b128(oA0[b], gA0 + kk);
    async_load_b128(oA1[b], gA1 + kk);
    async_load_b128(oW0[b], gW0 + kk);
    async_load_b128(oW1[b], gW1 + kk);
  };

  f32x8 acc[2][4];
#pragma unroll
  for (int i = 0; i < 2; ++i)
#pragma unroll
    for (int j = 0; j < 4; ++j)
      acc[i][j] = (f32x8){0.f,0.f,0.f,0.f,0.f,0.f,0.f,0.f};

  const int nk = K >> 5;
  issue(0, 0);
  for (int kt = 0; kt < nk; ++kt) {
    const int b = kt & 1;
    if (kt + 1 < nk) {
      issue(b ^ 1, (kt + 1) << 5);     // overlap next-tile DMA with this tile's WMMA
      WAIT_ASYNC(0x4);                 // in-order: previous tile's 4 loads are done
    } else {
      WAIT_ASYNC(0x0);
    }
    __syncthreads();

    FragBF16 af[2], bfm[4];
#pragma unroll
    for (int mt = 0; mt < 2; ++mt) {
      int r = wm * 32 + mt * 16 + lr;
      // A 16x32 bf16 layout: runs K[8kh..8kh+8) and K[16+8kh..16+8kh+8)
      af[mt].q[0] = *(uint4*)&As[b][r * 32 + kh * 8];
      af[mt].q[1] = *(uint4*)&As[b][r * 32 + 16 + kh * 8];
    }
#pragma unroll
    for (int nt = 0; nt < 4; ++nt) {
      int c = wn * 64 + nt * 16 + lr;
      // B 32x16 bf16 layout: runs K[8kh..8kh+8) and K[16+8kh..16+8kh+8)
      bfm[nt].q[0] = *(uint4*)&Bs[b][c * 32 + kh * 8];
      bfm[nt].q[1] = *(uint4*)&Bs[b][c * 32 + 16 + kh * 8];
    }
#pragma unroll
    for (int mt = 0; mt < 2; ++mt)
#pragma unroll
      for (int nt = 0; nt < 4; ++nt)
        acc[mt][nt] = __builtin_amdgcn_wmma_f32_16x16x32_bf16(
            false, af[mt].v, false, bfm[nt].v, (short)0, acc[mt][nt], false, false);
    __syncthreads();   // all waves done with buffer b before its next DMA overwrite
  }

  // epilogue: D layout: vgpr r, lanes0-15 -> m=r, lanes16-31 -> m=8+r; n=lane%16
#pragma unroll
  for (int mt = 0; mt < 2; ++mt)
#pragma unroll
    for (int nt = 0; nt < 4; ++nt) {
      int mb  = m0 + wm * 32 + mt * 16 + kh * 8;
      int col = n0 + wn * 64 + nt * 16 + lr;
      float bv = bias ? bias[col] : 0.f;
#pragma unroll
      for (int r = 0; r < 8; ++r) {
        float v = acc[mt][nt][r] + bv;
        if (act == 1)      v = v * (1.f / (1.f + __expf(-v)));   // silu
        else if (act == 2) v = 1.f / (1.f + __expf(-v));         // sigmoid
        size_t idx = (size_t)(mb + r) * N + col;
        if (Cf) Cf[idx] = v;
        if (Cb) Cb[idx] = to_bf16(v);
      }
    }
}

// ---------------------------------------------------------------------------
// Row LayerNorm (f32 in) -> bf16 out, optional broadcast row add AFTER LN.
// One block (128 threads) per row.
// ---------------------------------------------------------------------------
__global__ __launch_bounds__(128) void ln_rows_k(
    const float* __restrict__ X, const float* __restrict__ g, const float* __restrict__ b,
    const float* __restrict__ addrow, __bf16* __restrict__ outB, int D)
{
  __shared__ float s0[128], s1[128];
  long row = blockIdx.x;
  const float* xr = X + row * (long)D;
  float s = 0.f, ss = 0.f;
  for (int c = threadIdx.x; c < D; c += 128) { float v = xr[c]; s += v; ss += v * v; }
  s0[threadIdx.x] = s; s1[threadIdx.x] = ss;
  __syncthreads();
  for (int offl = 64; offl > 0; offl >>= 1) {
    if (threadIdx.x < offl) { s0[threadIdx.x] += s0[threadIdx.x + offl];
                              s1[threadIdx.x] += s1[threadIdx.x + offl]; }
    __syncthreads();
  }
  float mean = s0[0] / D;
  float var  = s1[0] / D - mean * mean;
  float rstd = rsqrtf(var + 1e-5f);
  for (int c = threadIdx.x; c < D; c += 128) {
    float v = (xr[c] - mean) * rstd * g[c] + b[c];
    if (addrow) v += addrow[c];
    outB[row * (long)D + c] = to_bf16(v);
  }
}

// ---------------------------------------------------------------------------
// Build context rows: [seg][t] -> retrieved(t<64) | persistent(t<68) | x(t<132)
// ---------------------------------------------------------------------------
__global__ void build_context_k(const __bf16* __restrict__ retr, const __bf16* __restrict__ pm,
                                const __bf16* __restrict__ xb, __bf16* __restrict__ ctx)
{
  const long total4 = 33792L * 128;  // uint4 chunks (8 bf16 each), 1024 cols/row
  for (long i = (long)blockIdx.x * blockDim.x + threadIdx.x; i < total4;
       i += (long)gridDim.x * blockDim.x) {
    long row = i >> 7; int j = (int)(i & 127);
    long seg = row / 132; int t = (int)(row % 132);
    const __bf16* src;
    if (t < 64)       src = &retr[(seg * 64 + t) * 1024];
    else if (t < 68)  src = &pm[(long)(t - 64) * 1024];
    else              src = &xb[(seg * 64 + (t - 68)) * 1024];
    *(uint4*)&ctx[row * 1024 + j * 8] = *(const uint4*)&src[j * 8];
  }
}

// ---------------------------------------------------------------------------
// Gated mix: out_bf16 = bf16(gate * attn + x)
// ---------------------------------------------------------------------------
__global__ void gate_mix_k(const float* __restrict__ gate, const float* __restrict__ attn,
                           const float* __restrict__ x, __bf16* __restrict__ out, long n)
{
  for (long i = (long)blockIdx.x * blockDim.x + threadIdx.x; i < n;
       i += (long)gridDim.x * blockDim.x)
    out[i] = to_bf16(gate[i] * attn[i] + x[i]);
}

// ---------------------------------------------------------------------------
// Attention: one block per (segment n, head h). S=64 queries, T=132 keys (pad 144
// for scores, 160 for the PV K-dim), hd=128.  4 waves; wave w owns query rows
// w*16..w*16+15.  Register-resident softmax using the WMMA D-tile layout.
// q/k staged via GLOBAL_LOAD_ASYNC_TO_LDS_B128 (ASYNCcnt).
// LDS phase aliasing (60KB total):
//   phase1: qs [64][128] @0 (16KB), ks [144][128] @16384 (36KB)
//   phase2: ps [64][160] @0 (20KB)
//   phase3: vt [128][160] @20480 (40KB)
// ---------------------------------------------------------------------------
#define TPAD 160
__global__ __launch_bounds__(128) void attention_k(
    const __bf16* __restrict__ qproj, const __bf16* __restrict__ kproj,
    const __bf16* __restrict__ vproj, __bf16* __restrict__ outb)
{
  __shared__ __attribute__((aligned(16))) char smem[61440];
  __bf16* qs = (__bf16*)smem;             // [64][128]
  __bf16* ks = (__bf16*)(smem + 16384);   // [144][128]
  __bf16* ps = (__bf16*)smem;             // [64][TPAD]
  __bf16* vt = (__bf16*)(smem + 20480);   // [128][TPAD]

  const int n = blockIdx.x, h = blockIdx.y;
  const int tid = threadIdx.x;
  const int wave = tid >> 5, lane = tid & 31, lr = lane & 15, kh = lane >> 4;
  const size_t qrow0 = (size_t)n * 64 * 1024 + h * 128;
  const size_t crow0 = (size_t)n * 132 * 1024 + h * 128;

  // ---- stage q (64x128) and k (132x128) via async DMA; zero-pad k rows 132..143
  for (int i = tid; i < 1024; i += 128) {            // 1024 x 16B chunks
    int r = i >> 3, j = i & 7;
    async_load_b128((unsigned)(size_t)&qs[r * 128 + j * 8],
                    &qproj[qrow0 + (size_t)r * 1024 + j * 8]);
  }
  for (int i = tid; i < 2112; i += 128) {            // 132 rows * 16 chunks
    int t = i >> 4, j = i & 15;
    async_load_b128((unsigned)(size_t)&ks[t * 128 + j * 8],
                    &kproj[crow0 + (size_t)t * 1024 + j * 8]);
  }
  for (int i = tid; i < 192; i += 128) {             // 12 pad rows * 16 chunks
    int t = 132 + (i >> 4), j = i & 15;
    uint4 z = {0u, 0u, 0u, 0u};
    *(uint4*)&ks[t * 128 + j * 8] = z;
  }
  WAIT_ASYNC(0x0);
  __syncthreads();

  // ---- phase 1: scores = q @ k^T * scale, 9 n-tiles kept in registers ----
  f32x8 sc[9];
  const float scale = 0.08838834764831845f;  // 1/sqrt(128)
#pragma unroll
  for (int nt = 0; nt < 9; ++nt) {
    f32x8 a8 = (f32x8){0.f,0.f,0.f,0.f,0.f,0.f,0.f,0.f};
#pragma unroll
    for (int kk = 0; kk < 128; kk += 32) {
      FragBF16 fa, fb;
      int qr = wave * 16 + lr;
      fa.q[0] = *(uint4*)&qs[qr * 128 + kk + kh * 8];
      fa.q[1] = *(uint4*)&qs[qr * 128 + kk + 16 + kh * 8];
      int tr = nt * 16 + lr;
      fb.q[0] = *(uint4*)&ks[tr * 128 + kk + kh * 8];
      fb.q[1] = *(uint4*)&ks[tr * 128 + kk + 16 + kh * 8];
      a8 = __builtin_amdgcn_wmma_f32_16x16x32_bf16(false, fa.v, false, fb.v,
                                                   (short)0, a8, false, false);
    }
    int t = nt * 16 + lr;
#pragma unroll
    for (int r = 0; r < 8; ++r)
      a8[r] = (t < 132) ? a8[r] * scale : -1e30f;
    sc[nt] = a8;
  }

  // ---- softmax in registers (rows live across 16-lane halves) ----
  float mx[8], sm[8];
#pragma unroll
  for (int r = 0; r < 8; ++r) {
    float m = sc[0][r];
#pragma unroll
    for (int nt = 1; nt < 9; ++nt) m = fmaxf(m, sc[nt][r]);
    m = fmaxf(m, __shfl_xor(m, 8, 32));
    m = fmaxf(m, __shfl_xor(m, 4, 32));
    m = fmaxf(m, __shfl_xor(m, 2, 32));
    m = fmaxf(m, __shfl_xor(m, 1, 32));
    mx[r] = m;
  }
#pragma unroll
  for (int r = 0; r < 8; ++r) {
    float s = 0.f;
#pragma unroll
    for (int nt = 0; nt < 9; ++nt) {
      float e = __expf(sc[nt][r] - mx[r]);
      sc[nt][r] = e; s += e;
    }
    s += __shfl_xor(s, 8, 32);
    s += __shfl_xor(s, 4, 32);
    s += __shfl_xor(s, 2, 32);
    s += __shfl_xor(s, 1, 32);
    sm[r] = 1.f / s;
  }
  __syncthreads();  // everyone done reading qs/ks -> safe to alias ps/vt

  // ---- write P (bf16) to LDS, zero pad cols [144,160) ----
#pragma unroll
  for (int nt = 0; nt < 9; ++nt)
#pragma unroll
    for (int r = 0; r < 8; ++r) {
      int row = wave * 16 + r + 8 * kh;
      int t   = nt * 16 + lr;
      ps[row * TPAD + t] = to_bf16(sc[nt][r] * sm[r]);
    }
  for (int i = lane; i < 256; i += 32) {
    int rr = i >> 4, cc = i & 15;
    ps[(wave * 16 + rr) * TPAD + 144 + cc] = to_bf16(0.f);
  }

  // ---- stage V transposed: vt[d][t], zero pad t in [132,160) ----
  for (int i = tid; i < 132 * 128; i += 128) {
    int t = i >> 7, d = i & 127;
    vt[d * TPAD + t] = vproj[crow0 + (size_t)t * 1024 + d];
  }
  for (int i = tid; i < 28 * 128; i += 128) {
    int t = 132 + (i >> 7), d = i & 127;
    vt[d * TPAD + t] = to_bf16(0.f);
  }
  __syncthreads();

  // ---- phase 3: attn = P @ V  (M=16/wave, N=128, K=160) ----
#pragma unroll
  for (int nt = 0; nt < 8; ++nt) {
    f32x8 a8 = (f32x8){0.f,0.f,0.f,0.f,0.f,0.f,0.f,0.f};
#pragma unroll
    for (int kk = 0; kk < TPAD; kk += 32) {
      FragBF16 fa, fb;
      int pr = wave * 16 + lr;
      fa.q[0] = *(uint4*)&ps[pr * TPAD + kk + kh * 8];
      fa.q[1] = *(uint4*)&ps[pr * TPAD + kk + 16 + kh * 8];
      int vr = nt * 16 + lr;  // output feature d
      fb.q[0] = *(uint4*)&vt[vr * TPAD + kk + kh * 8];
      fb.q[1] = *(uint4*)&vt[vr * TPAD + kk + 16 + kh * 8];
      a8 = __builtin_amdgcn_wmma_f32_16x16x32_bf16(false, fa.v, false, fb.v,
                                                   (short)0, a8, false, false);
    }
#pragma unroll
    for (int r = 0; r < 8; ++r) {
      int srow = wave * 16 + r + 8 * kh;
      int d    = nt * 16 + lr;
      outb[(size_t)(n * 64 + srow) * 1024 + h * 128 + d] = to_bf16(a8[r]);
    }
  }
}

// ---------------------------------------------------------------------------
// Host orchestration
// ---------------------------------------------------------------------------
extern "C" void kernel_launch(void* const* d_in, const int* in_sizes, int n_in,
                              void* d_out, int out_size, void* d_ws, size_t ws_size,
                              hipStream_t stream) {
  (void)in_sizes; (void)n_in; (void)out_size; (void)ws_size;

  const int NT = 16384, C = 1024, MM = 128;
  const long NCTX = 33792;  // 256 segments * 132 context tokens

  const float* x      = (const float*)d_in[0];
  const float* pm     = (const float*)d_in[1];
  const float* Wq     = (const float*)d_in[2];
  const float* Wk     = (const float*)d_in[3];
  const float* Wv     = (const float*)d_in[4];
  const float* memW1  = (const float*)d_in[5];
  const float* mem_b1 = (const float*)d_in[6];
  const float* ln1_g  = (const float*)d_in[7];
  const float* ln1_b  = (const float*)d_in[8];
  const float* memW2  = (const float*)d_in[9];
  const float* mem_b2 = (const float*)d_in[10];
  const float* ln2_g  = (const float*)d_in[11];
  const float* ln2_b  = (const float*)d_in[12];
  const float* memOW  = (const float*)d_in[13];
  const float* memOb  = (const float*)d_in[14];
  const float* mstate = (const float*)d_in[15];
  const float* mWq    = (const float*)d_in[16];
  const float* mbq    = (const float*)d_in[17];
  const float* mWk    = (const float*)d_in[18];
  const float* mbk    = (const float*)d_in[19];
  const float* mWv    = (const float*)d_in[20];
  const float* mbv    = (const float*)d_in[21];
  const float* mWo    = (const float*)d_in[22];
  const float* mbo    = (const float*)d_in[23];
  const float* gn_g   = (const float*)d_in[24];
  const float* gn_b   = (const float*)d_in[25];
  const float* gateW  = (const float*)d_in[26];
  const float* gate_b = (const float*)d_in[27];
  const float* outW   = (const float*)d_in[28];
  const float* out_b  = (const float*)d_in[29];

  char* base = (char*)d_ws;
  size_t off = 0;
  auto alloc = [&](size_t bytes) -> void* {
    void* p = base + off; off += (bytes + 255) & ~(size_t)255; return p;
  };

  // workspace layout (with phase aliasing)
  __bf16* xb     = (__bf16*)alloc((size_t)NT * C * 2);
  __bf16* wq_b   = (__bf16*)alloc((size_t)C * C * 2);
  __bf16* wk_b   = (__bf16*)alloc((size_t)C * C * 2);
  __bf16* wv_b   = (__bf16*)alloc((size_t)C * C * 2);
  __bf16* mWq_b  = (__bf16*)alloc((size_t)C * C * 2);
  __bf16* mWk_b  = (__bf16*)alloc((size_t)C * C * 2);
  __bf16* mWv_b  = (__bf16*)alloc((size_t)C * C * 2);
  __bf16* mWo_b  = (__bf16*)alloc((size_t)C * C * 2);
  __bf16* gW_b   = (__bf16*)alloc((size_t)C * C * 2);
  __bf16* oW_b   = (__bf16*)alloc((size_t)C * C * 2);
  __bf16* memW1b = (__bf16*)alloc((size_t)MM * C * 2);
  __bf16* memW2b = (__bf16*)alloc((size_t)MM * MM * 2);
  __bf16* memOWb = (__bf16*)alloc((size_t)C * MM * 2);
  __bf16* pm_b   = (__bf16*)alloc((size_t)4 * C * 2);
  float*  h1_f   = (float*)alloc((size_t)NT * MM * 4);
  __bf16* h1n_b  = (__bf16*)alloc((size_t)NT * MM * 2);
  float*  h2_f   = (float*)alloc((size_t)NT * MM * 4);
  __bf16* h2n_b  = (__bf16*)alloc((size_t)NT * MM * 2);

  void* S2  = alloc((size_t)NT * C * 2);    // qseg_b -> attnc_b
  void* S7  = alloc((size_t)NT * C * 2);    // retr_b -> qproj_b
  void* S8  = alloc((size_t)NCTX * C * 2);  // ctx_b  -> attno_f (f32, 64MB <= 66MB)
  void* S9  = alloc((size_t)NCTX * C * 2);  // kin/vin -> gln_b + mix_b
  void* S10 = alloc((size_t)NCTX * C * 2);  // kproj_b -> gate_f
  void* S11 = alloc((size_t)NCTX * C * 2);  // vproj_b

  __bf16* qseg_b  = (__bf16*)S2;
  __bf16* attnc_b = (__bf16*)S2;
  __bf16* retr_b  = (__bf16*)S7;
  __bf16* qproj_b = (__bf16*)S7;
  __bf16* ctx_b   = (__bf16*)S8;
  float*  attno_f = (float*)S8;
  __bf16* kin_b   = (__bf16*)S9;
  __bf16* vin_b   = (__bf16*)S9;
  __bf16* gln_b   = (__bf16*)S9;
  __bf16* mix_b   = (__bf16*)((char*)S9 + (size_t)NT * C * 2);
  __bf16* kproj_b = (__bf16*)S10;
  float*  gate_f  = (float*)S10;
  __bf16* vproj_b = (__bf16*)S11;

  auto cvt = [&](const float* src, __bf16* dst, long n) {
    cvt_f32_bf16_k<<<2048, 256, 0, stream>>>(src, dst, n);
  };
  auto gemm = [&](const __bf16* A, const __bf16* W_, const float* bias,
                  float* Cf, __bf16* Cb, int M_, int N_, int K_, int act) {
    dim3 g(M_ / 128, N_ / 128);
    gemm_bf16_wmma_k<<<g, 256, 0, stream>>>(A, W_, bias, Cf, Cb, M_, N_, K_, act);
  };

  // ---- convert inputs / weights to bf16 ----
  cvt(x, xb, (long)NT * C);
  cvt(Wq, wq_b, (long)C * C);   cvt(Wk, wk_b, (long)C * C);   cvt(Wv, wv_b, (long)C * C);
  cvt(mWq, mWq_b, (long)C * C); cvt(mWk, mWk_b, (long)C * C); cvt(mWv, mWv_b, (long)C * C);
  cvt(mWo, mWo_b, (long)C * C); cvt(gateW, gW_b, (long)C * C); cvt(outW, oW_b, (long)C * C);
  cvt(memW1, memW1b, (long)MM * C);
  cvt(memW2, memW2b, (long)MM * MM);
  cvt(memOW, memOWb, (long)C * MM);
  cvt(pm, pm_b, (long)4 * C);

  // ---- memory-retrieval path ----
  gemm(xb, wq_b, nullptr, nullptr, qseg_b, NT, C, C, 0);          // q_seg
  gemm(qseg_b, memW1b, mem_b1, h1_f, nullptr, NT, MM, C, 1);      // silu(W1 q + b1)
  ln_rows_k<<<NT, 128, 0, stream>>>(h1_f, ln1_g, ln1_b, nullptr, h1n_b, MM);
  gemm(h1n_b, memW2b, mem_b2, h2_f, nullptr, NT, MM, MM, 0);      // W2 h + b2
  ln_rows_k<<<NT, 128, 0, stream>>>(h2_f, ln2_g, ln2_b, mstate, h2n_b, MM);  // LN + state
  gemm(h2n_b, memOWb, memOb, nullptr, retr_b, NT, C, MM, 0);      // retrieved

  // ---- context + K/V/Q projections ----
  build_context_k<<<4096, 256, 0, stream>>>(retr_b, pm_b, xb, ctx_b);
  gemm(ctx_b, wk_b, nullptr, nullptr, kin_b, (int)NCTX, C, C, 0); // key_proj
  gemm(kin_b, mWk_b, mbk, nullptr, kproj_b, (int)NCTX, C, C, 0);  // mha K
  gemm(ctx_b, wv_b, nullptr, nullptr, vin_b, (int)NCTX, C, C, 0); // value_proj (reuses S9)
  gemm(vin_b, mWv_b, mbv, nullptr, vproj_b, (int)NCTX, C, C, 0);  // mha V
  gemm(qseg_b, mWq_b, mbq, nullptr, qproj_b, NT, C, C, 0);        // mha Q (reuses S7)

  // ---- attention (per segment x head) ----
  attention_k<<<dim3(256, 8), 128, 0, stream>>>(qproj_b, kproj_b, vproj_b, attnc_b);

  // ---- output projection + gate + final ----
  gemm(attnc_b, mWo_b, mbo, attno_f, nullptr, NT, C, C, 0);       // attn out proj
  ln_rows_k<<<NT, 128, 0, stream>>>(attno_f, gn_g, gn_b, nullptr, gln_b, C);
  gemm(gln_b, gW_b, gate_b, gate_f, nullptr, NT, C, C, 2);        // sigmoid gate
  gate_mix_k<<<4096, 256, 0, stream>>>(gate_f, attno_f, x, mix_b, (long)NT * C);
  gemm(mix_b, oW_b, out_b, (float*)d_out, nullptr, NT, C, C, 0);  // final out
}
